// RNNCell_81870666597039
// MI455X (gfx1250) — matrix operand order repaired and checked by reading, not compiled, not run on hardware
//
#include <hip/hip_runtime.h>
#include <hip/hip_fp16.h>
#include <math.h>

typedef _Float16 f16;
typedef __attribute__((ext_vector_type(16))) _Float16 v16h;
typedef __attribute__((ext_vector_type(8)))  float    v8f;

// --------------------------- CDNA5 async global->LDS -----------------------
// Use inline asm (bypasses clang builtin prototype quirks; see CDNA5 ISA
// GLOBAL_LOAD_ASYNC_TO_LDS_B128, tracked by ASYNCcnt).
#if defined(__gfx1250__)
#  define HAVE_ASYNC_LDS 1
#else
#  define HAVE_ASYNC_LDS 0
#endif

// 16B global->LDS copy: async DMA on CDNA5, sync fallback otherwise.
static __device__ __forceinline__ void copy16(const void* g, void* l) {
#if HAVE_ASYNC_LDS
    unsigned lds_addr = (unsigned)(unsigned long long)l;   // generic->LDS byte addr
    asm volatile("global_load_async_to_lds_b128 %0, %1, off"
                 :: "v"(lds_addr), "v"((unsigned long long)g)
                 : "memory");
#else
    *(uint4*)l = *(const uint4*)g;
#endif
}
static __device__ __forceinline__ void wait_async0() {
#if HAVE_ASYNC_LDS
    asm volatile("s_wait_asynccnt 0x0" ::: "memory");
#endif
}

static __device__ __forceinline__ v8f wmma_f16(v16h a, v16h b, v8f c) {
    // emits v_wmma_f32_16x16x32_f16
    return __builtin_amdgcn_wmma_f32_16x16x32_f16(false, a, false, b, (short)0, c, false, false);
}

// ---------------------------------------------------------------------------
// f32 -> f16 conversion (z and shared MLP weights)
// ---------------------------------------------------------------------------
__global__ __launch_bounds__(256)
void cvt_kernel(const float* __restrict__ s, f16* __restrict__ d, int n)
{
    int i = (blockIdx.x * 256 + threadIdx.x) * 4;
    if (i + 3 < n) {
        float4 v = *(const float4*)&s[i];
        d[i + 0] = (f16)v.x; d[i + 1] = (f16)v.y;
        d[i + 2] = (f16)v.z; d[i + 3] = (f16)v.w;
    } else {
        for (int k = 0; k < 4 && i + k < n; ++k) d[i + k] = (f16)s[i + k];
    }
}

// ---------------------------------------------------------------------------
// WMMA GEMM:  out = act(A[M,K] @ W[N,K]^T + bias)
//   flags bit0       : ReLU6 clip
//   flags bits[2:1]  : 0 = f16 row-major [M,N]
//                      1 = f16 transposed-in-batch: out[((m>>5)*N + n)*32 + (m&31)]
//                      2 = f32 row-major [M,N]
// Tile: 128(M) x 64(N), K-step 32, 256 threads = 8 waves (4x2 wave grid).
// Double-buffered LDS; next tile staged via async global->LDS DMA and
// overlapped with the current tile's WMMAs; one barrier per K-step.
// ---------------------------------------------------------------------------
__global__ __launch_bounds__(256)
void hyper_gemm(const f16* __restrict__ A, const f16* __restrict__ W,
                const float* __restrict__ bias, void* __restrict__ out,
                int M, int N, int K, int flags)
{
    __shared__ __attribute__((aligned(16))) f16 As[2][128][40]; // 40-half row pad
    __shared__ __attribute__((aligned(16))) f16 Bs[2][64][40];

    const int tid   = threadIdx.x;
    const int lane  = tid & 31;
    const int wave  = tid >> 5;
    const int wm    = (wave & 3) * 32;
    const int wn    = (wave >> 2) * 32;
    const int mBase = blockIdx.x * 128;
    const int n0    = blockIdx.y * 64;
    const bool relu6 = (flags & 1) != 0;
    const int  mode  = flags >> 1;

    // per-thread staging coordinates (one 16B chunk id per thread)
    const int sr  = tid >> 2;            // staging row 0..63
    const int sch = (tid & 3) * 8;       // staging half-offset {0,8,16,24}
    const f16* aP0 = A + (size_t)(mBase + sr) * K + sch;        // A rows sr, sr+64
    const f16* aP1 = aP0 + (size_t)64 * K;
    const bool bValid = (n0 + sr) < N;
    const f16* bP  = W + (size_t)(n0 + sr) * K + sch;

    // ragged-N rows: zero both buffers once, never stage them
    if (!bValid) {
        uint4 z = {0u, 0u, 0u, 0u};
        *(uint4*)&Bs[0][sr][sch] = z;
        *(uint4*)&Bs[1][sr][sch] = z;
    }

    auto stage = [&](int buf, int kk) {
        copy16(aP0 + kk, &As[buf][sr][sch]);
        copy16(aP1 + kk, &As[buf][64 + sr][sch]);
        if (bValid) copy16(bP + kk, &Bs[buf][sr][sch]);
    };

    const int row = lane & 15;
    const int kh  = (lane >> 4) * 8;   // A-frag K half-select (ISA 16-bit A 16x32)
    const int kh2 = (lane >> 4) * 16;  // unused: B frags use kh, same as A

    v8f c00 = {}, c01 = {}, c10 = {}, c11 = {};

    const int nk = K >> 5;
    stage(0, 0);
    for (int it = 0; it < nk; ++it) {
        wait_async0();            // this thread's DMA into buf (it&1) complete
        __syncthreads();          // everyone's tile visible; prior reads of nxt done
        if (it + 1 < nk)
            stage((it + 1) & 1, (it + 1) << 5);   // overlap DMA with WMMAs below

        const int buf = it & 1;
        union { v16h v; uint4 q[2]; } a0, a1, b0, b1;
        const f16* ar0 = &As[buf][wm + row][0];
        const f16* ar1 = &As[buf][wm + 16 + row][0];
        const f16* br0 = &Bs[buf][wn + row][0];
        const f16* br1 = &Bs[buf][wn + 16 + row][0];
        a0.q[0] = *(const uint4*)(ar0 + kh);   a0.q[1] = *(const uint4*)(ar0 + 16 + kh);
        a1.q[0] = *(const uint4*)(ar1 + kh);   a1.q[1] = *(const uint4*)(ar1 + 16 + kh);
        b0.q[0] = *(const uint4*)(br0 + kh);   b0.q[1] = *(const uint4*)(br0 + 16 + kh);
        b1.q[0] = *(const uint4*)(br1 + kh);   b1.q[1] = *(const uint4*)(br1 + 16 + kh);

        c00 = wmma_f16(a0.v, b0.v, c00);
        c01 = wmma_f16(a0.v, b1.v, c01);
        c10 = wmma_f16(a1.v, b0.v, c10);
        c11 = wmma_f16(a1.v, b1.v, c11);
    }

    // epilogue: C frag layout (VGPR r, lane L): M = r + 8*(L>>4), N = L&15
    auto store_frag = [&](const v8f& c, int mf, int nf) {
        int n = n0 + wn + nf * 16 + (lane & 15);
        if (n >= N) return;
        float bv = bias[n];
        int mb = mBase + wm + mf * 16 + 8 * (lane >> 4);
        #pragma unroll
        for (int r = 0; r < 8; ++r) {
            float v = c[r] + bv;
            if (relu6) v = fminf(fmaxf(v, 0.0f), 6.0f);
            int m = mb + r;
            if (mode == 0)
                ((f16*)out)[(size_t)m * N + n] = (f16)v;
            else if (mode == 1)
                ((f16*)out)[(((size_t)(m >> 5)) * N + n) * 32 + (m & 31)] = (f16)v;
            else
                ((float*)out)[(size_t)m * N + n] = v;
        }
    };
    store_frag(c00, 0, 0);
    store_frag(c01, 0, 1);
    store_frag(c10, 1, 0);
    store_frag(c11, 1, 1);
}

// ---------------------------------------------------------------------------
// Final bh_c layer: N == 1  ->  per-row dot product (K = 64)
// ---------------------------------------------------------------------------
__global__ __launch_bounds__(256)
void rowdot_kernel(const f16* __restrict__ A, const f16* __restrict__ w,
                   const float* __restrict__ bias, float* __restrict__ out, int K)
{
    int m = blockIdx.x * 256 + threadIdx.x;
    const f16* a = A + (size_t)m * K;
    float acc = bias[0];
    #pragma unroll 8
    for (int k = 0; k < 64; ++k) acc = fmaf((float)a[k], (float)w[k], acc);
    out[m] = acc;
}

// ---------------------------------------------------------------------------
// Sequential RNN scan: one workgroup (128 threads) per batch element.
// W_hh (128x128 f32) in LDS padded to 129 floats/row (bank-conflict free).
// W_ih row (32 f16) in registers. x staged in double-buffered 64-step tiles
// via async global->LDS DMA (next tile hidden under 64 sequential steps);
// h history staged in LDS and flushed as coalesced float4 runs.
// ---------------------------------------------------------------------------
__global__ __launch_bounds__(128)
void rnn_scan(const float* __restrict__ x, const float* __restrict__ h0,
              const f16* __restrict__ Wih, const float* __restrict__ Whh,
              const float* __restrict__ bh, float* __restrict__ out)
{
    extern __shared__ char smem[];
    float* Ws   = (float*)smem;        // 128*129
    float* xs0  = Ws + 128 * 129;      // 2 x 32*64 (double buffer)
    float* hcur = xs0 + 2 * 32 * 64;   // 128
    float* ob   = hcur + 128;          // 128*64

    const int b = blockIdx.x;
    const int j = threadIdx.x;

    for (int i = j; i < 128 * 128; i += 128)
        Ws[(i >> 7) * 129 + (i & 127)] = Whh[(size_t)b * 16384 + i];

    float wi[32];
    const f16* wp = Wih + (size_t)b * 4096 + (size_t)j * 32;
    #pragma unroll
    for (int k = 0; k < 32; ++k) wi[k] = (float)wp[k];

    const float bj = bh[b * 128 + j];
    hcur[j] = h0[b * 128 + j];

    const float* xb = x + (size_t)b * 32 * 512;
    float* op = out + (size_t)b * 128 * 512 + (size_t)j * 512;

    // stage a 32x64 x-tile (512 16B chunks, 4 per thread)
    auto stage_x = [&](float* dst, int t0) {
        #pragma unroll
        for (int q = 0; q < 4; ++q) {
            int fi = (j + 128 * q) * 4;           // float index in tile
            int r = fi >> 6, col = fi & 63;
            copy16(xb + (size_t)r * 512 + t0 + col, dst + r * 64 + col);
        }
    };

    stage_x(xs0, 0);
    for (int ci = 0; ci < 8; ++ci) {
        wait_async0();
        __syncthreads();                           // tile + (first iter) Ws/hcur visible
        if (ci + 1 < 8)
            stage_x(xs0 + ((ci + 1) & 1) * 2048, (ci + 1) * 64);  // hidden under 64 steps
        const float* xsc = xs0 + (ci & 1) * 2048;
        const int t0 = ci * 64;

        for (int tt = 0; tt < 64; ++tt) {
            float acc = bj;
            #pragma unroll 8
            for (int i2 = 0; i2 < 32; ++i2)
                acc = fmaf(wi[i2], xsc[i2 * 64 + tt], acc);   // broadcast reads
            const float* wr = Ws + j * 129;
            #pragma unroll 8
            for (int k = 0; k < 128; ++k)
                acc = fmaf(hcur[k], wr[k], acc);              // conflict-free
            float hn = tanhf(acc);
            __syncthreads();                                   // all h reads done
            hcur[j] = hn;
            ob[j * 64 + tt] = hn;
            __syncthreads();                                   // h visible
        }
        #pragma unroll
        for (int tt = 0; tt < 64; tt += 4)                     // coalesced flush
            *(float4*)&op[t0 + tt] = *(const float4*)&ob[j * 64 + tt];
    }
}

// ---------------------------------------------------------------------------
// Host orchestration
// ---------------------------------------------------------------------------
extern "C" void kernel_launch(void* const* d_in, const int* in_sizes, int n_in,
                              void* d_out, int out_size, void* d_ws, size_t ws_size,
                              hipStream_t stream)
{
    (void)in_sizes; (void)n_in; (void)out_size; (void)ws_size;
    const float* x  = (const float*)d_in[0];
    const float* h0 = (const float*)d_in[1];
    const float* z  = (const float*)d_in[2];

    struct L { int wi, bi, N, K; size_t off; };
    L wih_s[8], wih_c[8], whh_s[8], whh_c[8], bh_s[3], bh_c[3];
    size_t woff = 0;
    auto fillS8 = [&](L* a, int base) {
        for (int i = 0; i < 8; ++i) {
            a[i] = { base + 2 * i, base + 2 * i + 1, (i == 7) ? 128 : 256, 256, woff };
            woff += (size_t)a[i].N * a[i].K;
        }
    };
    auto fillC8 = [&](L* a, int base, int lastN) {
        for (int i = 0; i < 8; ++i) {
            a[i] = { base + 2 * i, base + 2 * i + 1,
                     (i == 7) ? lastN : 64, (i == 0) ? 32 : 64, woff };
            woff += (size_t)a[i].N * a[i].K;
        }
    };
    fillS8(wih_s, 3);
    fillC8(wih_c, 19, 32);
    fillS8(whh_s, 35);
    fillC8(whh_c, 51, 128);
    for (int i = 0; i < 3; ++i) {
        bh_s[i] = { 67 + 2 * i, 68 + 2 * i, (i == 2) ? 128 : 256, 256, woff };
        woff += (size_t)bh_s[i].N * bh_s[i].K;
    }
    for (int i = 0; i < 3; ++i) {
        bh_c[i] = { 73 + 2 * i, 74 + 2 * i, (i == 2) ? 1 : 64, (i == 0) ? 32 : 64, woff };
        woff += (size_t)bh_c[i].N * bh_c[i].K;
    }

    // scratch carve
    char* ws = (char*)d_ws;
    auto up = [](size_t v) { return (v + 255) & ~(size_t)255; };
    f16* wf16 = (f16*)ws;
    size_t cur = up(woff * sizeof(f16));
    f16*   z16   = (f16*)(ws + cur); cur += up((size_t)32768 * 256 * 2);
    f16*   T0    = (f16*)(ws + cur); cur += up((size_t)32768 * 256 * 2);
    f16*   T1    = (f16*)(ws + cur); cur += up((size_t)32768 * 256 * 2);
    f16*   Wih16 = (f16*)(ws + cur); cur += up((size_t)1024 * 128 * 32 * 2);
    float* Whh32 = (float*)(ws + cur); cur += up((size_t)1024 * 128 * 128 * 4);
    float* bh32  = (float*)(ws + cur);

    auto cvt = [&](const float* s, f16* d, size_t n) {
        int grid = (int)(((n + 3) / 4 + 255) / 256);
        cvt_kernel<<<grid, 256, 0, stream>>>(s, d, (int)n);
    };
    cvt(z, z16, (size_t)32768 * 256);
    auto cvtChain = [&](const L* a, int n) {
        for (int i = 0; i < n; ++i)
            cvt((const float*)d_in[a[i].wi], wf16 + a[i].off, (size_t)a[i].N * a[i].K);
    };
    cvtChain(wih_s, 8); cvtChain(wih_c, 8);
    cvtChain(whh_s, 8); cvtChain(whh_c, 8);
    cvtChain(bh_s, 3);  cvtChain(bh_c, 3);

    auto gemm = [&](const f16* Ain, const L& l, void* outp, int M, int flags) {
        dim3 g(M / 128, (l.N + 63) / 64);
        hyper_gemm<<<g, 256, 0, stream>>>(Ain, wf16 + l.off,
                                          (const float*)d_in[l.bi],
                                          outp, M, l.N, l.K, flags);
    };

    // flags: bit0 relu6; bits[2:1] outmode (0 f16, 1 f16-transposed, 2 f32)
    auto runChain = [&](const L* sL, int nS, const L* cL, int nC,
                        int finalFlags, void* finalOut) {
        const f16* curA = z16; f16* pp[2] = { T0, T1 }; int p = 0;
        for (int i = 0; i < nS; ++i) {
            int flags = (i < nS - 1) ? 1 : (1 << 1); // relu6 | transposed, no act
            gemm(curA, sL[i], pp[p], 32768, flags);
            curA = pp[p]; p ^= 1;
        }
        for (int i = 0; i < nC; ++i) {
            bool last = (i == nC - 1);
            gemm(curA, cL[i], last ? finalOut : (void*)pp[p],
                 131072, last ? finalFlags : 1);
            curA = pp[p]; p ^= 1;
        }
    };

    runChain(wih_s, 8, wih_c, 8, 0 /*f16 row-major (B,128,32)*/, Wih16);
    runChain(whh_s, 8, whh_c, 8, (2 << 1) /*f32 row-major (B,128,128)*/, Whh32);

    { // bh chain: 3 s-layers, 2 WMMA c-layers, final N=1 row-dot
        const f16* curA = z16; f16* pp[2] = { T0, T1 }; int p = 0;
        for (int i = 0; i < 3; ++i) {
            int flags = (i < 2) ? 1 : (1 << 1);
            gemm(curA, bh_s[i], pp[p], 32768, flags);
            curA = pp[p]; p ^= 1;
        }
        for (int i = 0; i < 2; ++i) {
            gemm(curA, bh_c[i], pp[p], 131072, 1);
            curA = pp[p]; p ^= 1;
        }
        rowdot_kernel<<<131072 / 256, 256, 0, stream>>>(
            curA, wf16 + bh_c[2].off, (const float*)d_in[bh_c[2].bi], bh32, 64);
    }

    size_t smem = (size_t)(128 * 129 + 2 * 32 * 64 + 128 + 128 * 64) * sizeof(float);
    rnn_scan<<<1024, 128, smem, stream>>>(x, h0, Wih16, Whh32, bh32, (float*)d_out);
}
